// TRILIXTransformer_34729105555793
// MI455X (gfx1250) — matrix operand order repaired and checked by reading, not compiled, not run on hardware
//
#include <hip/hip_runtime.h>
#include <hip/hip_bf16.h>
#include <math.h>

typedef _Float16 half_t;
typedef __attribute__((ext_vector_type(8)))  _Float16 v8h;
typedef __attribute__((ext_vector_type(16))) _Float16 v16h;
typedef __attribute__((ext_vector_type(8)))  float    v8f;

union V16 { v16h v; v8h h[2]; };

// ---------------- model constants ----------------
constexpr int SEQ   = 2048;
constexpr int HID   = 1024;
constexpr int NLAY  = 4;
constexpr int NHEAD = 16;
constexpr int NKVH  = 4;
constexpr int HDIM  = 64;
constexpr int FFD   = 4096;
constexpr int RNK   = 128;
constexpr int NCODE = 256;
constexpr int NEXP  = 8;
constexpr int VOC   = 32000;
constexpr float BETAC = 0.25f;

// ---------------- WMMA helper ----------------
__device__ inline v8f wmma32(const V16& a, const V16& b, v8f c) {
  return __builtin_amdgcn_wmma_f32_16x16x32_f16(
      /*neg_a=*/false, a.v, /*neg_b=*/false, b.v,
      /*c_mod=*/(short)0, c, /*reuse_a=*/false, /*reuse_b=*/false);
}

// =====================================================================
// Generic WMMA GEMM:  C[M,N](f32) = A[M,K](f16, row-major) * BT[N,K]^T
// One wave per block computes a 32x64 tile (2 M-subtiles x 4 N-subtiles,
// 8 WMMAs per 32-deep k-step; B-fragments shared across both A-fragments).
// grid = (N/64, M/32), block = 32.
// A-frag (16-bit A 16x32): lanes 0-15 row M=l, K k0..k0+15;
//                          lanes16-31 row M=l-16, K k0+16..k0+31.
// B-frag (16-bit B 32x16): lane l holds column n=l%16,
//                          K = k0 + (l<16 ? 0..15 : 16..31).
// =====================================================================
__global__ void gemm_wmma(const half_t* __restrict__ A,
                          const half_t* __restrict__ BT,
                          float* __restrict__ C, int M, int N, int K) {
  const int lane = threadIdx.x & 31;
  const int hf   = lane >> 4;
  const int l16  = lane & 15;
  const int m0   = blockIdx.y * 32;
  const int n0   = blockIdx.x * 64;
  v8f a0n0 = {}, a0n1 = {}, a0n2 = {}, a0n3 = {};
  v8f a1n0 = {}, a1n1 = {}, a1n2 = {}, a1n3 = {};
  const half_t* arow0 = A + (size_t)(m0 + l16) * K + hf * 16;
  const half_t* arow1 = A + (size_t)(m0 + 16 + l16) * K + hf * 16;
  for (int k0 = 0; k0 < K; k0 += 32) {
    V16 af0, af1;
    af0.h[0] = *(const v8h*)(arow0 + k0);
    af0.h[1] = *(const v8h*)(arow0 + k0 + 8);
    af1.h[0] = *(const v8h*)(arow1 + k0);
    af1.h[1] = *(const v8h*)(arow1 + k0 + 8);
    const half_t* bbase = BT + k0 + hf * 16;
    V16 b0, b1, b2, b3;
    const half_t* p0 = bbase + (size_t)(n0 +  0 + l16) * K;
    const half_t* p1 = bbase + (size_t)(n0 + 16 + l16) * K;
    const half_t* p2 = bbase + (size_t)(n0 + 32 + l16) * K;
    const half_t* p3 = bbase + (size_t)(n0 + 48 + l16) * K;
    b0.h[0] = *(const v8h*)(p0); b0.h[1] = *(const v8h*)(p0 + 8);
    b1.h[0] = *(const v8h*)(p1); b1.h[1] = *(const v8h*)(p1 + 8);
    b2.h[0] = *(const v8h*)(p2); b2.h[1] = *(const v8h*)(p2 + 8);
    b3.h[0] = *(const v8h*)(p3); b3.h[1] = *(const v8h*)(p3 + 8);
    a0n0 = wmma32(af0, b0, a0n0);
    a1n0 = wmma32(af1, b0, a1n0);
    a0n1 = wmma32(af0, b1, a0n1);
    a1n1 = wmma32(af1, b1, a1n1);
    a0n2 = wmma32(af0, b2, a0n2);
    a1n2 = wmma32(af1, b2, a1n2);
    a0n3 = wmma32(af0, b3, a0n3);
    a1n3 = wmma32(af1, b3, a1n3);
  }
  // C layout: element (vgpr v, lane) -> row m0 + v + 8*hf, col n0 + t*16 + l16
#pragma unroll
  for (int v = 0; v < 8; ++v) {
    float* crow0 = C + (size_t)(m0 + v + 8 * hf) * N + n0 + l16;
    crow0[ 0] = a0n0[v];
    crow0[16] = a0n1[v];
    crow0[32] = a0n2[v];
    crow0[48] = a0n3[v];
    float* crow1 = C + (size_t)(m0 + 16 + v + 8 * hf) * N + n0 + l16;
    crow1[ 0] = a1n0[v];
    crow1[16] = a1n1[v];
    crow1[32] = a1n2[v];
    crow1[48] = a1n3[v];
  }
}

// =====================================================================
// Flash attention: one wave per (head, 16-row q tile). GQA 16q/4kv, hd=64.
// q16:(s, h, d) f16 roped ; k16:(s, kvh, d) f16 roped ; vt16:(kvh, d, s) f16
// out:(s, h, d) f32
// =====================================================================
__global__ void flash_attn(const half_t* __restrict__ q16,
                           const half_t* __restrict__ k16,
                           const half_t* __restrict__ vt16,
                           float* __restrict__ outp) {
  const int h    = blockIdx.y;
  const int q0   = blockIdx.x * 16;
  const int kvh  = h / (NHEAD / NKVH);
  const int lane = threadIdx.x & 31;
  const int hf   = lane >> 4;
  const int l16  = lane & 15;
  __shared__ __align__(16) half_t lds_p[16 * 32];

  v8f o0 = {}, o1 = {}, o2 = {}, o3 = {};
  float mrow[8], lrow[8];
#pragma unroll
  for (int v = 0; v < 8; ++v) { mrow[v] = -3.0e38f; lrow[v] = 0.f; }

  // Q A-fragments for head-dim chunks [0,32) and [32,64)
  V16 qa0, qa1;
  const half_t* qrow = q16 + (size_t)(q0 + l16) * (NHEAD * HDIM) + h * HDIM + hf * 16;
  qa0.h[0] = *(const v8h*)(qrow +  0);
  qa0.h[1] = *(const v8h*)(qrow + 8);
  qa1.h[0] = *(const v8h*)(qrow + 32);
  qa1.h[1] = *(const v8h*)(qrow + 40);

  const float scale = 0.125f; // 1/sqrt(64)

  for (int kb = 0; kb <= q0 + 15; kb += 32) {
    // ---- scores: two 16x16 key tiles ----
    v8f sc0 = {}, sc1 = {};
    {
      int key = kb + l16; if (key > SEQ - 1) key = SEQ - 1;
      const half_t* kr = k16 + (size_t)key * (NKVH * HDIM) + kvh * HDIM;
      V16 b0, b1;
      b0.h[0] = *(const v8h*)(kr + hf * 16);
      b0.h[1] = *(const v8h*)(kr + hf * 16 + 8);
      b1.h[0] = *(const v8h*)(kr + 32 + hf * 16);
      b1.h[1] = *(const v8h*)(kr + 32 + hf * 16 + 8);
      sc0 = wmma32(qa0, b0, sc0);
      sc0 = wmma32(qa1, b1, sc0);
    }
    {
      int key = kb + 16 + l16; if (key > SEQ - 1) key = SEQ - 1;
      const half_t* kr = k16 + (size_t)key * (NKVH * HDIM) + kvh * HDIM;
      V16 b0, b1;
      b0.h[0] = *(const v8h*)(kr + hf * 16);
      b0.h[1] = *(const v8h*)(kr + hf * 16 + 8);
      b1.h[0] = *(const v8h*)(kr + 32 + hf * 16);
      b1.h[1] = *(const v8h*)(kr + 32 + hf * 16 + 8);
      sc1 = wmma32(qa0, b0, sc1);
      sc1 = wmma32(qa1, b1, sc1);
    }
    // ---- online softmax (row = q0 + v + 8*hf, cols spread over 16 lanes) ----
    float p0v[8], p1v[8];
#pragma unroll
    for (int v = 0; v < 8; ++v) {
      const int row = q0 + v + 8 * hf;
      const int c0 = kb + l16, c1 = kb + 16 + l16;
      float a0 = (c0 <= row) ? sc0[v] * scale : -3.0e38f;
      float a1 = (c1 <= row) ? sc1[v] * scale : -3.0e38f;
      float mx = fmaxf(a0, a1);
      for (int m = 1; m <= 8; m <<= 1) mx = fmaxf(mx, __shfl_xor(mx, m, 32));
      const float mnew = fmaxf(mrow[v], mx);
      const float p0 = __expf(a0 - mnew);
      const float p1 = __expf(a1 - mnew);
      float rs = p0 + p1;
      for (int m = 1; m <= 8; m <<= 1) rs += __shfl_xor(rs, m, 32);
      const float corr = __expf(mrow[v] - mnew);
      lrow[v] = lrow[v] * corr + rs;
      mrow[v] = mnew;
      o0[v] *= corr; o1[v] *= corr; o2[v] *= corr; o3[v] *= corr;
      p0v[v] = p0; p1v[v] = p1;
    }
    // ---- convert P from C-layout to A-layout via LDS ----
#pragma unroll
    for (int v = 0; v < 8; ++v) {
      lds_p[(v + 8 * hf) * 32 + l16]      = (half_t)p0v[v];
      lds_p[(v + 8 * hf) * 32 + 16 + l16] = (half_t)p1v[v];
    }
    __syncthreads();
    V16 pa;
    pa.h[0] = *(const v8h*)(&lds_p[l16 * 32 + hf * 16]);
    pa.h[1] = *(const v8h*)(&lds_p[l16 * 32 + 8 + hf * 16]);
    __syncthreads();
    // ---- O += P @ V (4 head-dim tiles of 16) ----
    int ks = kb + hf * 16; if (ks > SEQ - 16) ks = SEQ - 16;
    const half_t* vb = vt16 + (size_t)kvh * HDIM * SEQ + (size_t)l16 * SEQ + ks;
    V16 vb0, vb1, vb2, vb3;
    vb0.h[0] = *(const v8h*)(vb + (size_t)0 * 16 * SEQ);
    vb0.h[1] = *(const v8h*)(vb + (size_t)0 * 16 * SEQ + 8);
    vb1.h[0] = *(const v8h*)(vb + (size_t)1 * 16 * SEQ);
    vb1.h[1] = *(const v8h*)(vb + (size_t)1 * 16 * SEQ + 8);
    vb2.h[0] = *(const v8h*)(vb + (size_t)2 * 16 * SEQ);
    vb2.h[1] = *(const v8h*)(vb + (size_t)2 * 16 * SEQ + 8);
    vb3.h[0] = *(const v8h*)(vb + (size_t)3 * 16 * SEQ);
    vb3.h[1] = *(const v8h*)(vb + (size_t)3 * 16 * SEQ + 8);
    o0 = wmma32(pa, vb0, o0);
    o1 = wmma32(pa, vb1, o1);
    o2 = wmma32(pa, vb2, o2);
    o3 = wmma32(pa, vb3, o3);
  }
  // ---- normalize and store ----
#pragma unroll
  for (int v = 0; v < 8; ++v) {
    const float invl = 1.f / lrow[v];
    float* orow = outp + (size_t)(q0 + v + 8 * hf) * (NHEAD * HDIM) + h * HDIM + l16;
    orow[ 0] = o0[v] * invl;
    orow[16] = o1[v] * invl;
    orow[32] = o2[v] * invl;
    orow[48] = o3[v] * invl;
  }
}

// =====================================================================
// VQ + top-2 routing. One 256-thread block per token.
// z:(S,R) f32, Cw:(256,128) f32, Rw:(128,8) f32.
// Writes ze:(S, E*R) f16 with ze[s, e*R+r] = w_e * code[r] (0 elsewhere),
// accumulates commit loss into aux.
// =====================================================================
__global__ void vq_route(const float* __restrict__ z, const float* __restrict__ Cw,
                         const float* __restrict__ Rw, half_t* __restrict__ ze,
                         float* __restrict__ aux) {
  __shared__ float zsh[RNK];
  __shared__ float dsh[NCODE];
  __shared__ int   idxsh[NCODE];
  __shared__ float code_sh[RNK];
  __shared__ float vsh[NEXP];
  __shared__ int   e0s, e1s;
  __shared__ float w0s, w1s;
  const int s = blockIdx.x, t = threadIdx.x;
  if (t < RNK) zsh[t] = z[(size_t)s * RNK + t];
  __syncthreads();
  // distance to code t:  |c|^2 - 2 z.c  (|z|^2 constant over codes)
  {
    const float* c = Cw + (size_t)t * RNK;
    float dot = 0.f, cn = 0.f;
    for (int r = 0; r < RNK; ++r) { float cv = c[r]; dot += zsh[r] * cv; cn += cv * cv; }
    dsh[t] = cn - 2.f * dot;
    idxsh[t] = t;
  }
  __syncthreads();
  for (int off = NCODE / 2; off > 0; off >>= 1) {
    if (t < off) {
      if (dsh[t + off] < dsh[t]) { dsh[t] = dsh[t + off]; idxsh[t] = idxsh[t + off]; }
    }
    __syncthreads();
  }
  const int best = idxsh[0];
  if (t < RNK) code_sh[t] = Cw[(size_t)best * RNK + t];
  __syncthreads();
  // commit loss: BETA * mean((z - code)^2) over (S*R)
  {
    float d = (t < RNK) ? (zsh[t] - code_sh[t]) : 0.f;
    dsh[t] = d * d;
    __syncthreads();
    for (int off = NCODE / 2; off > 0; off >>= 1) {
      if (t < off) dsh[t] += dsh[t + off];
      __syncthreads();
    }
    if (t == 0) atomicAdd(aux, BETAC * dsh[0] / (float)((size_t)SEQ * RNK));
  }
  // routing: vals[e] = code . Rw[:,e]
  if (t < NEXP) {
    float acc = 0.f;
    for (int r = 0; r < RNK; ++r) acc += code_sh[r] * Rw[r * NEXP + t];
    vsh[t] = acc;
  }
  __syncthreads();
  if (t == 0) {
    int e0 = 0; float b0 = -3.0e38f;
    for (int e = 0; e < NEXP; ++e) if (vsh[e] > b0) { b0 = vsh[e]; e0 = e; }
    int e1 = 0; float b1 = -3.0e38f;
    for (int e = 0; e < NEXP; ++e) if (e != e0 && vsh[e] > b1) { b1 = vsh[e]; e1 = e; }
    const float w0 = 1.f / (1.f + __expf(b1 - b0));
    e0s = e0; e1s = e1; w0s = w0; w1s = 1.f - w0;
  }
  __syncthreads();
  // write expert-scaled z row (dense E*R layout, zeros for inactive experts)
  for (int idx = t; idx < NEXP * RNK; idx += blockDim.x) {
    const int e = idx >> 7, r = idx & (RNK - 1);
    float val = 0.f;
    if (e == e0s) val = w0s * code_sh[r];
    else if (e == e1s) val = w1s * code_sh[r];
    ze[(size_t)s * (NEXP * RNK) + idx] = (half_t)val;
  }
}

// ---------------- small elementwise / utility kernels ----------------
__global__ void embed_k(const int* __restrict__ ids, const float* __restrict__ emb,
                        float* __restrict__ h) {
  int idx = blockIdx.x * blockDim.x + threadIdx.x;
  if (idx >= SEQ * HID) return;
  int s = idx / HID, j = idx - s * HID;
  h[idx] = emb[(size_t)ids[s] * HID + j];
}

__global__ void rmsnorm_k(const float* __restrict__ x, const float* __restrict__ w,
                          half_t* __restrict__ out, int D) {
  __shared__ float red[256];
  const int row = blockIdx.x;
  const float* xr = x + (size_t)row * D;
  float s = 0.f;
  for (int i = threadIdx.x; i < D; i += blockDim.x) { float v = xr[i]; s += v * v; }
  red[threadIdx.x] = s;
  __syncthreads();
  for (int off = 128; off > 0; off >>= 1) {
    if ((int)threadIdx.x < off) red[threadIdx.x] += red[threadIdx.x + off];
    __syncthreads();
  }
  const float inv = 1.f / sqrtf(red[0] / (float)D + 1e-5f);
  for (int i = threadIdx.x; i < D; i += blockDim.x)
    out[(size_t)row * D + i] = (half_t)(xr[i] * inv * w[i]);
}

// transpose + f32->f16:  in (K x N) row-major  ->  out (N x K) row-major
__global__ void cvtT_k(const float* __restrict__ in, half_t* __restrict__ out,
                       int K, int N) {
  int idx = blockIdx.x * blockDim.x + threadIdx.x;
  if (idx >= K * N) return;
  int n = idx / K, k = idx - n * K;
  out[idx] = (half_t)in[(size_t)k * N + n];
}

__global__ void cvt_k(const float* __restrict__ in, half_t* __restrict__ out, int n) {
  int i = blockIdx.x * blockDim.x + threadIdx.x;
  if (i < n) out[i] = (half_t)in[i];
}

__global__ void add_k(float* __restrict__ a, const float* __restrict__ b, int n) {
  int i = blockIdx.x * blockDim.x + threadIdx.x;
  if (i < n) a[i] += b[i];
}

__global__ void silu_mul_k(const float* __restrict__ g, const float* __restrict__ u,
                           half_t* __restrict__ out, int n) {
  int i = blockIdx.x * blockDim.x + threadIdx.x;
  if (i < n) {
    float x = g[i];
    out[i] = (half_t)((x / (1.f + __expf(-x))) * u[i]);
  }
}

// RoPE + f32->f16 for (s, nh, 64) layout
__global__ void rope_k(const float* __restrict__ x, half_t* __restrict__ out, int nh) {
  const int s = blockIdx.x;
  const int D = nh * HDIM;
  const float lgth = logf(500000.f);
  for (int i = threadIdx.x; i < D; i += blockDim.x) {
    const int d = i & (HDIM - 1);
    const int fi = d & (HDIM / 2 - 1);            // freq index d % 32
    const float invf = __expf(-((float)(2 * fi) / (float)HDIM) * lgth);
    const float ang = (float)s * invf;
    const float c = cosf(ang), sn = sinf(ang);
    const float xv = x[(size_t)s * D + i];
    const float xr = (d & 1) ? x[(size_t)s * D + i - 1] : -x[(size_t)s * D + i + 1];
    out[(size_t)s * D + i] = (half_t)(xv * c + xr * sn);
  }
}

// v (s, kvh, d) f32 -> vt (kvh, d, s) f16
__global__ void vtrans_k(const float* __restrict__ v, half_t* __restrict__ vt) {
  int idx = blockIdx.x * blockDim.x + threadIdx.x;
  if (idx >= NKVH * HDIM * SEQ) return;
  int s = idx % SEQ;
  int d = (idx / SEQ) % HDIM;
  int hh = idx / (SEQ * HDIM);
  vt[idx] = (half_t)v[(size_t)s * (NKVH * HDIM) + hh * HDIM + d];
}

__global__ void zero1_k(float* p) { p[0] = 0.f; }

// =====================================================================
// Host-side orchestration
// =====================================================================
extern "C" void kernel_launch(void* const* d_in, const int* in_sizes, int n_in,
                              void* d_out, int out_size, void* d_ws, size_t ws_size,
                              hipStream_t stream) {
  (void)in_sizes; (void)n_in; (void)out_size; (void)ws_size;

  const int*   ids        = (const int*)d_in[0];
  /* d_in[1] = attention_mask (unused; causal mask computed analytically) */
  const float* embed      = (const float*)d_in[2];
  const float* lm_head    = (const float*)d_in[3];
  const float* final_norm = (const float*)d_in[4];
  const float* ln1        = (const float*)d_in[5];
  const float* ln2        = (const float*)d_in[6];

  // projections in PROJ order: q, k, v, o, gate, up, down -> (A, C, R, B) each
  const float *pA[7], *pC[7], *pR[7], *pB[7];
  for (int p = 0; p < 7; ++p) {
    pA[p] = (const float*)d_in[7 + p * 4 + 0];
    pC[p] = (const float*)d_in[7 + p * 4 + 1];
    pR[p] = (const float*)d_in[7 + p * 4 + 2];
    pB[p] = (const float*)d_in[7 + p * 4 + 3];
  }
  static const int DOU[7] = {NHEAD * HDIM, NKVH * HDIM, NKVH * HDIM, HID, FFD, FFD, HID};

  // -------- workspace layout --------
  char* wsb = (char*)d_ws;
  size_t off = 0;
  auto alloc = [&](size_t bytes) -> void* {
    void* p = wsb + off;
    off = (off + bytes + 255) & ~(size_t)255;
    return p;
  };
  half_t* wA    = (half_t*)alloc((size_t)RNK * FFD * 2);          // transposed A weight (<=128x4096)
  half_t* wB    = (half_t*)alloc((size_t)FFD * (NEXP * RNK) * 2); // transposed B weight (<=4096x1024)
  float*  h     = (float*) alloc((size_t)SEQ * HID * 4);
  half_t* hn16  = (half_t*)alloc((size_t)SEQ * HID * 2);
  float*  zbuf  = (float*) alloc((size_t)SEQ * RNK * 4);
  half_t* zebuf = (half_t*)alloc((size_t)SEQ * NEXP * RNK * 2);
  float*  qf    = (float*) alloc((size_t)SEQ * NHEAD * HDIM * 4);
  float*  kf    = (float*) alloc((size_t)SEQ * NKVH * HDIM * 4);
  float*  vf    = (float*) alloc((size_t)SEQ * NKVH * HDIM * 4);
  half_t* q16   = (half_t*)alloc((size_t)SEQ * NHEAD * HDIM * 2);
  half_t* k16   = (half_t*)alloc((size_t)SEQ * NKVH * HDIM * 2);
  half_t* vt16  = (half_t*)alloc((size_t)NKVH * HDIM * SEQ * 2);
  float*  att   = (float*) alloc((size_t)SEQ * HID * 4);
  half_t* x16   = (half_t*)alloc((size_t)SEQ * FFD * 2);          // f16 GEMM-A input (attn out / ffn in)
  float*  yA    = (float*) alloc((size_t)SEQ * FFD * 4);          // gate / generic y
  float*  yB    = (float*) alloc((size_t)SEQ * FFD * 4);          // up
  half_t* lmT   = (half_t*)alloc((size_t)VOC * HID * 2);          // lm_head transposed f16

  float* logits = (float*)d_out;
  float* auxp   = (float*)d_out + (size_t)SEQ * VOC;

  auto cdiv = [](long long a, long long b) { return (unsigned)((a + b - 1) / b); };

  // TRILIX: xin (S x Kin, f16) -> yout (S x DOU[p], f32)
  auto trilix = [&](const half_t* xin, int Kin, int p, int l, float* yout) {
    // z = xin @ A[l]           (A: Kin x R)
    cvtT_k<<<cdiv((long long)Kin * RNK, 256), 256, 0, stream>>>(
        pA[p] + (size_t)l * Kin * RNK, wA, Kin, RNK);
    gemm_wmma<<<dim3(RNK / 64, SEQ / 32), 32, 0, stream>>>(xin, wA, zbuf, SEQ, RNK, Kin);
    // VQ + routing -> ze (dense E*R)
    vq_route<<<SEQ, NCODE, 0, stream>>>(
        zbuf, pC[p] + (size_t)l * NCODE * RNK, pR[p] + (size_t)l * RNK * NEXP, zebuf, auxp);
    // y = ze @ B_flat[l]       (B_flat: (E*R) x DOU)
    cvtT_k<<<cdiv((long long)(NEXP * RNK) * DOU[p], 256), 256, 0, stream>>>(
        pB[p] + (size_t)l * NEXP * RNK * DOU[p], wB, NEXP * RNK, DOU[p]);
    gemm_wmma<<<dim3(DOU[p] / 64, SEQ / 32), 32, 0, stream>>>(
        zebuf, wB, yout, SEQ, DOU[p], NEXP * RNK);
  };

  // -------- forward pass --------
  zero1_k<<<1, 1, 0, stream>>>(auxp);
  embed_k<<<cdiv((long long)SEQ * HID, 256), 256, 0, stream>>>(ids, embed, h);

  for (int l = 0; l < NLAY; ++l) {
    // attention block
    rmsnorm_k<<<SEQ, 256, 0, stream>>>(h, ln1 + (size_t)l * HID, hn16, HID);
    trilix(hn16, HID, 0, l, qf);   // q
    trilix(hn16, HID, 1, l, kf);   // k
    trilix(hn16, HID, 2, l, vf);   // v
    rope_k<<<SEQ, 256, 0, stream>>>(qf, q16, NHEAD);
    rope_k<<<SEQ, 256, 0, stream>>>(kf, k16, NKVH);
    vtrans_k<<<cdiv((long long)NKVH * HDIM * SEQ, 256), 256, 0, stream>>>(vf, vt16);
    flash_attn<<<dim3(SEQ / 16, NHEAD), 32, 0, stream>>>(q16, k16, vt16, att);
    cvt_k<<<cdiv((long long)SEQ * HID, 256), 256, 0, stream>>>(att, x16, SEQ * HID);
    trilix(x16, HID, 3, l, yA);    // o
    add_k<<<cdiv((long long)SEQ * HID, 256), 256, 0, stream>>>(h, yA, SEQ * HID);

    // FFN block
    rmsnorm_k<<<SEQ, 256, 0, stream>>>(h, ln2 + (size_t)l * HID, hn16, HID);
    trilix(hn16, HID, 4, l, yA);   // gate
    trilix(hn16, HID, 5, l, yB);   // up
    silu_mul_k<<<cdiv((long long)SEQ * FFD, 256), 256, 0, stream>>>(yA, yB, x16, SEQ * FFD);
    trilix(x16, FFD, 6, l, qf);    // down (output S x HID, reuse qf)
    add_k<<<cdiv((long long)SEQ * HID, 256), 256, 0, stream>>>(h, qf, SEQ * HID);
  }

  // final norm + lm_head
  rmsnorm_k<<<SEQ, 256, 0, stream>>>(h, final_norm, hn16, HID);
  cvtT_k<<<cdiv((long long)HID * VOC, 256), 256, 0, stream>>>(lm_head, lmT, HID, VOC);
  gemm_wmma<<<dim3(VOC / 64, SEQ / 32), 32, 0, stream>>>(hn16, lmT, logits, SEQ, VOC, HID);
}
